// diff_attention_67070209294502
// MI455X (gfx1250) — matrix operand order repaired and checked by reading, not compiled, not run on hardware
//
#include <hip/hip_runtime.h>

#define B_DIM 8
#define S_DIM 2048
#define D_DIM 1024

typedef __attribute__((ext_vector_type(16))) __bf16 v16bf;
typedef __attribute__((ext_vector_type(8)))  float  v8f;

union FragU { uint4 q[2]; unsigned short s[16]; v16bf v; };

__device__ __forceinline__ unsigned short f2bf(float x) {
  union { float f; unsigned u; } v; v.f = x;
  unsigned r = v.u + 0x7fffu + ((v.u >> 16) & 1u);   // round-to-nearest-even
  return (unsigned short)(r >> 16);
}

__device__ __forceinline__ v8f wmma_bf16(v16bf a, v16bf b, v8f c) {
  // 8 args: (neg_a, A, neg_b, B, c_mod, C, reuse_a, reuse_b)
  return __builtin_amdgcn_wmma_f32_16x16x32_bf16(false, a, false, b, (short)0, c, false, false);
}

// ---- A fragment (16x32 bf16, row-major source, ld = row stride) ----
// lane<16: row=row0+lane,    K = {k0..k0+7, k0+16..k0+23}
// lane>=16: row=row0+lane-16, K = {k0+8..k0+15, k0+24..k0+31}
__device__ __forceinline__ v16bf load_fragA_bf16(const unsigned short* __restrict__ base,
                                                 int ld, int row0, int k0) {
  int lane = threadIdx.x & 31;
  const unsigned short* p = base + (size_t)(row0 + (lane & 15)) * (size_t)ld
                                 + (k0 + ((lane >> 4) << 3));
  FragU f;
  f.q[0] = *(const uint4*)(p);
  f.q[1] = *(const uint4*)(p + 16);
  return f.v;
}

// ---- B fragment (32x16 bf16). B[k][n] = M[n][k] with M row-major, ld = K stride ----
// lane<16: col n=n0+lane,    K = {k0..k0+7, k0+16..k0+23}
// lane>=16: col n=n0+lane-16, K = {k0+8..k0+15, k0+24..k0+31}
__device__ __forceinline__ v16bf load_fragB_bf16(const unsigned short* __restrict__ base,
                                                 int ld, int n0, int k0) {
  int lane = threadIdx.x & 31;
  const unsigned short* p = base + (size_t)(n0 + (lane & 15)) * (size_t)ld
                                 + (k0 + ((lane >> 4) << 3));
  FragU f;
  f.q[0] = *(const uint4*)(p);
  f.q[1] = *(const uint4*)(p + 16);
  return f.v;
}

// =====================================================================
// Kernel 0: transpose + convert Wq2, Wk2, Wv[D:2D] to bf16 [out][in]
// =====================================================================
__global__ void prep_weights(const float* __restrict__ Wq2, const float* __restrict__ Wk2,
                             const float* __restrict__ Wv,
                             unsigned short* __restrict__ Wq2t,
                             unsigned short* __restrict__ Wk2t,
                             unsigned short* __restrict__ Wv2t) {
  int idx = blockIdx.x * blockDim.x + threadIdx.x;   // 0 .. D*D-1
  int i = idx >> 10;         // input row (in-dim)
  int j = idx & 1023;        // input col (out-dim), coalesced read
  const float* src; unsigned short* dst;
  if (blockIdx.y == 0)      { src = Wq2;                        dst = Wq2t; }
  else if (blockIdx.y == 1) { src = Wk2;                        dst = Wk2t; }
  else                      { src = Wv + (size_t)D_DIM * D_DIM; dst = Wv2t; }
  dst[(size_t)j * D_DIM + i] = f2bf(src[idx]);
}

// =====================================================================
// Kernel 1: X2 -> bf16 (one conversion per element, vectorized 4-wide)
// =====================================================================
__global__ void prep_x2b(const float* __restrict__ X2, unsigned int* __restrict__ X2b) {
  size_t idx = (size_t)blockIdx.x * blockDim.x + threadIdx.x;  // one float4 each
  float4 c = *(const float4*)(X2 + idx * 4);
  unsigned lo = (unsigned)f2bf(c.x) | ((unsigned)f2bf(c.y) << 16);
  unsigned hi = (unsigned)f2bf(c.z) | ((unsigned)f2bf(c.w) << 16);
  X2b[idx * 2]     = lo;
  X2b[idx * 2 + 1] = hi;
}

// =====================================================================
// Kernel 2: v1[b][n] = X1[b] . Wv[0:D, n] + bv[n]   (per-batch V bias)
// =====================================================================
__global__ void prep_v1(const float* __restrict__ X1, const float* __restrict__ Wv,
                        const float* __restrict__ bv, float* __restrict__ v1) {
  int n = blockIdx.x * blockDim.x + threadIdx.x;   // 0..D-1
  int b = blockIdx.y;
  float acc = bv[n];
  for (int d = 0; d < D_DIM; ++d)
    acc += X1[b * D_DIM + d] * Wv[(size_t)d * D_DIM + n];
  v1[b * D_DIM + n] = acc;
}

// =====================================================================
// Kernel 3: fused projection GEMMs (all-bf16 operands).
//   op0: Q2 = X2b @ Wq2 + bq2   -> bf16 [S,D]
//   op1: K2 = X2b @ Wk2 + bk2   -> bf16 [S,D]
//   op2: V  = X2b @ Wv2 + v1[b] -> stored transposed Vt bf16 [D,S]
// Block: 256 thr = 8 waves (2x4); block tile 64x256; wave tile 32x64
// (2 A-frags x 4 B-frags -> 8 WMMA accs, 1.5 b128 loads per WMMA).
// =====================================================================
__global__ __launch_bounds__(256) void proj_gemm(
    const unsigned short* __restrict__ X2b,
    const unsigned short* __restrict__ Wq2t, const unsigned short* __restrict__ Wk2t,
    const unsigned short* __restrict__ Wv2t,
    const float* __restrict__ bq2, const float* __restrict__ bk2,
    const float* __restrict__ v1,
    unsigned short* __restrict__ Q2, unsigned short* __restrict__ K2,
    unsigned short* __restrict__ Vt) {
  int b    = blockIdx.y;
  int op   = blockIdx.z;
  int wave = threadIdx.x >> 5;
  int lane = threadIdx.x & 31;
  int wm = wave & 1, wn = wave >> 1;
  int m0 = (blockIdx.x & 31) * 64 + wm * 32;   // S rows   (32 m-blocks)
  int n0 = (blockIdx.x >> 5) * 256 + wn * 64;  // D cols   (4 n-blocks)
  const unsigned short* W  = (op == 0) ? Wq2t : (op == 1) ? Wk2t : Wv2t;
  const unsigned short* Xb = X2b + (size_t)b * S_DIM * D_DIM;

  v8f acc[2][4];
  for (int i = 0; i < 2; ++i)
    for (int j = 0; j < 4; ++j)
      for (int e = 0; e < 8; ++e) acc[i][j][e] = 0.f;

  for (int kc = 0; kc < D_DIM; kc += 32) {
    v16bf a0 = load_fragA_bf16(Xb, D_DIM, m0, kc);
    v16bf a1 = load_fragA_bf16(Xb, D_DIM, m0 + 16, kc);
    #pragma unroll
    for (int j = 0; j < 4; ++j) {
      v16bf bf = load_fragB_bf16(W, D_DIM, n0 + j * 16, kc);
      acc[0][j] = wmma_bf16(a0, bf, acc[0][j]);
      acc[1][j] = wmma_bf16(a1, bf, acc[1][j]);
    }
  }

  int hi = lane >> 4, nl = lane & 15;
  for (int j = 0; j < 4; ++j) {
    int n = n0 + j * 16 + nl;
    float bias = (op == 0) ? bq2[n] : (op == 1) ? bk2[n] : v1[b * D_DIM + n];
    #pragma unroll
    for (int i = 0; i < 2; ++i) {
      #pragma unroll
      for (int r = 0; r < 8; ++r) {
        int m = m0 + i * 16 + r + 8 * hi;
        unsigned short val = f2bf(acc[i][j][r] + bias);
        if (op == 0)      Q2[(size_t)b * S_DIM * D_DIM + (size_t)m * D_DIM + n] = val;
        else if (op == 1) K2[(size_t)b * S_DIM * D_DIM + (size_t)m * D_DIM + n] = val;
        else              Vt[(size_t)b * D_DIM * S_DIM + (size_t)n * S_DIM + m] = val;
      }
    }
  }
}

// =====================================================================
// Kernel 4: scores + softmax + alpha correction, P = softmax(Q2 K2^T * s) - a/S.
// Block = 16 queries x full 2048 keys; wave w owns keys [w*256, w*256+256).
// Row stats: shfl_xor lane reduce (16-lane groups) + LDS cross-wave reduce.
// =====================================================================
__global__ __launch_bounds__(256) void scores_softmax(
    const unsigned short* __restrict__ Q2, const unsigned short* __restrict__ K2,
    const float* __restrict__ alpha, unsigned short* __restrict__ P) {
  __shared__ float smax[8][16];
  __shared__ float ssum[8][16];
  int b    = blockIdx.y;
  int q0   = blockIdx.x * 16;
  int wave = threadIdx.x >> 5;
  int lane = threadIdx.x & 31;
  int hi = lane >> 4, nl = lane & 15;
  int kb0 = wave * 256;
  const unsigned short* Qb = Q2 + (size_t)b * S_DIM * D_DIM;
  const unsigned short* Kb = K2 + (size_t)b * S_DIM * D_DIM;

  v8f acc[16];
  for (int t = 0; t < 16; ++t)
    for (int i = 0; i < 8; ++i) acc[t][i] = 0.f;

  for (int kc = 0; kc < D_DIM; kc += 32) {
    v16bf a = load_fragA_bf16(Qb, D_DIM, q0, kc);
    #pragma unroll
    for (int t = 0; t < 16; ++t) {
      v16bf bf = load_fragB_bf16(Kb, D_DIM, kb0 + t * 16, kc);
      acc[t] = wmma_bf16(a, bf, acc[t]);
    }
  }

  // per-wave row max over this wave's 256 keys (raw scores; scale folded in exp)
  float rmax[8];
  #pragma unroll
  for (int r = 0; r < 8; ++r) {
    float m = acc[0][r];
    #pragma unroll
    for (int t = 1; t < 16; ++t) m = fmaxf(m, acc[t][r]);
    for (int d = 1; d <= 8; d <<= 1) m = fmaxf(m, __shfl_xor(m, d, 32));
    rmax[r] = m;
  }
  if (nl == 0)
    for (int r = 0; r < 8; ++r) smax[wave][r + 8 * hi] = rmax[r];
  __syncthreads();

  float gmax[8];
  #pragma unroll
  for (int r = 0; r < 8; ++r) {
    int row = r + 8 * hi;
    float m = smax[0][row];
    for (int w = 1; w < 8; ++w) m = fmaxf(m, smax[w][row]);
    gmax[r] = m;
  }

  const float scale = 0.03125f;   // 1/sqrt(1024)
  float rsum[8];
  #pragma unroll
  for (int r = 0; r < 8; ++r) rsum[r] = 0.f;
  for (int t = 0; t < 16; ++t) {
    #pragma unroll
    for (int r = 0; r < 8; ++r) {
      float p = __expf((acc[t][r] - gmax[r]) * scale);
      acc[t][r] = p;
      rsum[r] += p;
    }
  }
  #pragma unroll
  for (int r = 0; r < 8; ++r) {
    float s = rsum[r];
    for (int d = 1; d <= 8; d <<= 1) s += __shfl_xor(s, d, 32);
    rsum[r] = s;
  }
  if (nl == 0)
    for (int r = 0; r < 8; ++r) ssum[wave][r + 8 * hi] = rsum[r];
  __syncthreads();

  float aOverS = alpha[0] * (1.0f / (float)S_DIM);
  unsigned short* Pb = P + (size_t)b * S_DIM * S_DIM;
  #pragma unroll
  for (int r = 0; r < 8; ++r) {
    int row = r + 8 * hi;
    float s = ssum[0][row];
    for (int w = 1; w < 8; ++w) s += ssum[w][row];
    float invl = 1.0f / s;
    for (int t = 0; t < 16; ++t)
      Pb[(size_t)(q0 + row) * S_DIM + (kb0 + t * 16 + nl)] = f2bf(acc[t][r] * invl - aOverS);
  }
}

// =====================================================================
// Kernel 5: O = P @ V  (P bf16 [S,S], V via Vt bf16 [D,S]) -> f32 out [B,S,D]
// Same 2x4-wave / 32x64-wave-tile structure as proj_gemm.
// =====================================================================
__global__ __launch_bounds__(256) void pv_gemm(const unsigned short* __restrict__ P,
                                               const unsigned short* __restrict__ Vt,
                                               float* __restrict__ out) {
  int b    = blockIdx.y;
  int wave = threadIdx.x >> 5;
  int lane = threadIdx.x & 31;
  int wm = wave & 1, wn = wave >> 1;
  int m0 = (blockIdx.x & 31) * 64 + wm * 32;   // S rows
  int n0 = (blockIdx.x >> 5) * 256 + wn * 64;  // D cols
  const unsigned short* Pb = P  + (size_t)b * S_DIM * S_DIM;
  const unsigned short* Vb = Vt + (size_t)b * D_DIM * S_DIM;

  v8f acc[2][4];
  for (int i = 0; i < 2; ++i)
    for (int j = 0; j < 4; ++j)
      for (int e = 0; e < 8; ++e) acc[i][j][e] = 0.f;

  for (int kc = 0; kc < S_DIM; kc += 32) {
    if (kc + 64 < S_DIM)   // gfx1250 global_prefetch of next A strip
      __builtin_prefetch(Pb + (size_t)(m0 + (threadIdx.x & 15)) * S_DIM + kc + 64, 0, 3);
    v16bf a0 = load_fragA_bf16(Pb, S_DIM, m0, kc);
    v16bf a1 = load_fragA_bf16(Pb, S_DIM, m0 + 16, kc);
    #pragma unroll
    for (int j = 0; j < 4; ++j) {
      v16bf bf = load_fragB_bf16(Vb, S_DIM, n0 + j * 16, kc);
      acc[0][j] = wmma_bf16(a0, bf, acc[0][j]);
      acc[1][j] = wmma_bf16(a1, bf, acc[1][j]);
    }
  }

  int hi = lane >> 4, nl = lane & 15;
  for (int j = 0; j < 4; ++j) {
    int n = n0 + j * 16 + nl;
    #pragma unroll
    for (int i = 0; i < 2; ++i) {
      #pragma unroll
      for (int r = 0; r < 8; ++r) {
        int m = m0 + i * 16 + r + 8 * hi;
        out[(size_t)b * S_DIM * D_DIM + (size_t)m * D_DIM + n] = acc[i][j][r];
      }
    }
  }
}

// =====================================================================
extern "C" void kernel_launch(void* const* d_in, const int* in_sizes, int n_in,
                              void* d_out, int out_size, void* d_ws, size_t ws_size,
                              hipStream_t stream) {
  (void)in_sizes; (void)n_in; (void)out_size; (void)ws_size;
  const float* X1    = (const float*)d_in[0];
  const float* X2    = (const float*)d_in[1];
  // d_in[2],[3],[6],[7] (Wq1,bq1,Wk1,bk1) are algebraically eliminated:
  // softmax of a row-constant matrix is uniform 1/S.
  const float* Wq2   = (const float*)d_in[4];
  const float* bq2   = (const float*)d_in[5];
  const float* Wk2   = (const float*)d_in[8];
  const float* bk2   = (const float*)d_in[9];
  const float* Wv    = (const float*)d_in[10];
  const float* bv    = (const float*)d_in[11];
  const float* alpha = (const float*)d_in[12];
  float* out = (float*)d_out;

  // Workspace layout (~199 MB total):
  //   Q2, K2 : bf16 [B,S,D]  (32 MB each)
  //   Vt     : bf16 [B,D,S]  (32 MB)
  //   P      : bf16 [B,S,S]  (64 MB)
  //   X2b    : bf16 [B,S,D]  (32 MB)
  //   Wq2t/Wk2t/Wv2t : bf16 [D,D] (2 MB each)
  //   v1     : f32 [B,D]
  const size_t SD = (size_t)S_DIM * D_DIM;       // 2M elems
  unsigned short* Q2   = (unsigned short*)d_ws;
  unsigned short* K2   = Q2 + (size_t)B_DIM * SD;
  unsigned short* Vt   = K2 + (size_t)B_DIM * SD;
  unsigned short* Pm   = Vt + (size_t)B_DIM * SD;
  unsigned short* X2b  = Pm + (size_t)B_DIM * S_DIM * S_DIM;
  unsigned short* Wq2t = X2b + (size_t)B_DIM * SD;
  unsigned short* Wk2t = Wq2t + (size_t)D_DIM * D_DIM;
  unsigned short* Wv2t = Wk2t + (size_t)D_DIM * D_DIM;
  float*          v1   = (float*)(Wv2t + (size_t)D_DIM * D_DIM);

  prep_weights<<<dim3((D_DIM * D_DIM) / 256, 3), 256, 0, stream>>>(
      Wq2, Wk2, Wv, Wq2t, Wk2t, Wv2t);
  prep_x2b<<<dim3((B_DIM * S_DIM * D_DIM) / (256 * 4)), 256, 0, stream>>>(
      X2, (unsigned int*)X2b);
  prep_v1<<<dim3(D_DIM / 256, B_DIM), 256, 0, stream>>>(X1, Wv, bv, v1);
  proj_gemm<<<dim3((S_DIM / 64) * (D_DIM / 256), B_DIM, 3), 256, 0, stream>>>(
      X2b, Wq2t, Wk2t, Wv2t, bq2, bk2, v1, Q2, K2, Vt);
  scores_softmax<<<dim3(S_DIM / 16, B_DIM), 256, 0, stream>>>(Q2, K2, alpha, Pm);
  pv_gemm<<<dim3((S_DIM / 64) * (D_DIM / 256), B_DIM), 256, 0, stream>>>(Pm, Vt, out);
}
